// PGN_71528385348322
// MI455X (gfx1250) — matrix operand order repaired and checked by reading, not compiled, not run on hardware
//
#include <hip/hip_runtime.h>
#include <hip/hip_bf16.h>

// ---------------------------------------------------------------------------
// PGN message-passing layer for MI455X (gfx1250, wave32, WMMA).
//   B=4, N=512, H=128, MID=OUT=128, Z=256.
// Fusion: agg[b,i,m] = max(-1e6, c1[b,i,m] + max_{j:adj[b,j,i]} (m2[b,j,m] +
//                          edge[b,j,i,:]@W_me[:,m]))
//   c1 = z@W_m1+b_m1 + g@W_mg+b_mg + b_me   (j-invariant terms hoisted)
// Dominant GEMM in f16 WMMA with f32 accumulate (memory-bound path);
// adjacency mask applied as additive {0,-INF} penalties (no cmp/cndmask).
// ---------------------------------------------------------------------------

#define BB   4
#define NN   512
#define HH   128
#define MIDD 128
#define IT   16          // i-tile (WMMA M)
#define JC   128         // j-chunk per workgroup
#define NJ   (NN / JC)   // 4 j-chunks
#define NIT  (NN / IT)   // 32 i-tiles

typedef _Float16 v16h __attribute__((ext_vector_type(16)));
typedef _Float16 h8   __attribute__((ext_vector_type(8)));
typedef float    v8f  __attribute__((ext_vector_type(8)));

// ---------------------------------------------------------------------------
// Kernel 1: fp32 precompute of m2, c1, o1  (tiny GEMMs over z = [node|hidden])
// ---------------------------------------------------------------------------
__global__ __launch_bounds__(256) void pgn_pre_kernel(
    const float* __restrict__ node, const float* __restrict__ hidden,
    const float* __restrict__ graph,
    const float* __restrict__ W_m1, const float* __restrict__ b_m1,
    const float* __restrict__ W_m2, const float* __restrict__ b_m2,
    const float* __restrict__ W_mg, const float* __restrict__ b_mg,
    const float* __restrict__ b_me,
    const float* __restrict__ W_o1, const float* __restrict__ b_o1,
    float* __restrict__ m2, float* __restrict__ c1, float* __restrict__ o1)
{
    const int idx = blockIdx.x * 256 + threadIdx.x;       // (b*N+n)*128 + m
    const int m   = idx & (MIDD - 1);
    const size_t bn = (size_t)(idx >> 7);                 // b*N + n
    const int b   = (int)(bn >> 9);

    const float* nrow = node   + bn * HH;
    const float* hrow = hidden + bn * HH;
    const float* grow = graph  + (size_t)b * HH;

    float s1 = 0.f, s2 = 0.f, s3 = 0.f, sg = 0.f;
#pragma unroll 8
    for (int k = 0; k < HH; ++k) {
        const float zk = nrow[k];
        s1 = fmaf(zk, W_m1[k * MIDD + m], s1);
        s2 = fmaf(zk, W_m2[k * MIDD + m], s2);
        s3 = fmaf(zk, W_o1[k * MIDD + m], s3);
    }
#pragma unroll 8
    for (int k = 0; k < HH; ++k) {
        const float zk = hrow[k];
        s1 = fmaf(zk, W_m1[(k + HH) * MIDD + m], s1);
        s2 = fmaf(zk, W_m2[(k + HH) * MIDD + m], s2);
        s3 = fmaf(zk, W_o1[(k + HH) * MIDD + m], s3);
    }
#pragma unroll 8
    for (int k = 0; k < HH; ++k)
        sg = fmaf(grow[k], W_mg[k * MIDD + m], sg);

    m2[idx] = s2 + b_m2[m];
    c1[idx] = s1 + b_m1[m] + sg + b_mg[m] + b_me[m];
    o1[idx] = s3 + b_o1[m];
}

// ---------------------------------------------------------------------------
// Kernel 2: dominant edge GEMM + masked max over a j-chunk, via WMMA f16.
// Grid: B * NIT * NJ workgroups; 128 threads = 4 wave32 waves.
// Wave w owns output columns [32w, 32w+32) as two 16-col WMMA B fragments.
// ---------------------------------------------------------------------------
__global__ __launch_bounds__(128) void pgn_edge_wmma_kernel(
    const float* __restrict__ edge, const int* __restrict__ adj,
    const float* __restrict__ W_me, const float* __restrict__ m2,
    float* __restrict__ part)
{
    const int blk  = blockIdx.x;
    const int jc   = blk % NJ;
    const int it   = (blk / NJ) % NIT;
    const int b    = blk / (NJ * NIT);
    const int i0   = it * IT;
    const int j0   = jc * JC;

    const int tid  = threadIdx.x;
    const int lane = tid & 31;
    const int wave = tid >> 5;

    __shared__ _Float16 Atile[2][IT * HH];   // 2 x 4 KB f16 tiles (dbl buffer)
    __shared__ float    pen[JC * IT];        // additive mask: 0.0 or -INF

    // --- adj penalties for this (b, i-tile, j-chunk): one j row per thread --
    {
        const int* ar = adj + ((size_t)b * NN + (j0 + tid)) * NN + i0;
#pragma unroll
        for (int r = 0; r < IT; ++r)
            pen[tid * IT + r] = (ar[r] != 0) ? 0.0f : -INFINITY;
    }

    // --- resident B fragments: W_me[:, 32w .. 32w+31] as f16 ---------------
    // 16-bit B 32x16 layout (ISA 7.12.2): lane<16 holds K=e (e<8) / 8+e (e>=8),
    // col=lane; lane>=16 holds K=8+e / 16+e, col=lane-16. Chunk c adds 32c to K.
    union BF { v16h v; _Float16 e[16]; };
    BF bf[2][4];
    {
        const int col0 = wave * 32 + (lane & 15);
        const int kb   = (lane < 16) ? 0 : 16;
#pragma unroll
        for (int t = 0; t < 2; ++t)
#pragma unroll
            for (int c = 0; c < 4; ++c)
#pragma unroll
                for (int e = 0; e < 16; ++e)
                    bf[t][c].e[e] =
                        (_Float16)W_me[(32 * c + (kb >> 1) + e + (e & 8)) * MIDD + col0 + 16 * t];
    }

    // --- cooperative fill: 16x128 fp32 edge tile -> f16 in LDS -------------
    // For fixed (b,j) the [i0:i0+16, 0:128] tile is 2048 contiguous floats.
    auto fill = [&](int j, int buf) {
        const float* src =
            edge + (((size_t)b * NN + j) * NN + i0) * (size_t)HH + tid * 16;
        const float4* s4 = (const float4*)src;
        float4 f0 = s4[0], f1 = s4[1], f2 = s4[2], f3 = s4[3];
        h8 lo, hi;
        lo[0] = (_Float16)f0.x; lo[1] = (_Float16)f0.y;
        lo[2] = (_Float16)f0.z; lo[3] = (_Float16)f0.w;
        lo[4] = (_Float16)f1.x; lo[5] = (_Float16)f1.y;
        lo[6] = (_Float16)f1.z; lo[7] = (_Float16)f1.w;
        hi[0] = (_Float16)f2.x; hi[1] = (_Float16)f2.y;
        hi[2] = (_Float16)f2.z; hi[3] = (_Float16)f2.w;
        hi[4] = (_Float16)f3.x; hi[5] = (_Float16)f3.y;
        hi[6] = (_Float16)f3.z; hi[7] = (_Float16)f3.w;
        *(h8*)&Atile[buf][tid * 16]     = lo;
        *(h8*)&Atile[buf][tid * 16 + 8] = hi;
    };

    fill(j0, 0);
    __syncthreads();

    float acc0[8], acc1[8];
#pragma unroll
    for (int r = 0; r < 8; ++r) { acc0[r] = -INFINITY; acc1[r] = -INFINITY; }

    const int row   = lane & 15;
    const int kbase = (lane < 16) ? 0 : 8;        // 16-bit A layout (7.12.2)
    const int hioff = (lane < 16) ? 0 : 8;        // f32 C/D layout row offset
    const int mlane = wave * 32 + (lane & 15);

    for (int jj = 0; jj < JC; ++jj) {
        const int j   = j0 + jj;
        const int buf = jj & 1;
        if (jj + 1 < JC) fill(j + 1, buf ^ 1);

        const float m2a = m2[((size_t)b * NN + j) * MIDD + mlane];
        const float m2b = m2[((size_t)b * NN + j) * MIDD + mlane + 16];

        // this lane's 8 row penalties (rows hioff..hioff+7), 32B aligned
        const float4* pp = (const float4*)&pen[jj * IT + hioff];
        const float4 pa = pp[0], pb = pp[1];
        const float pen8[8] = {pa.x, pa.y, pa.z, pa.w, pb.x, pb.y, pb.z, pb.w};

        v8f d0 = {}, d1 = {};
#pragma unroll
        for (int c = 0; c < 4; ++c) {
            const _Float16* ab = &Atile[buf][row * HH + 32 * c + kbase];
            h8 r1 = *(const h8*)(ab);        // K = 32c+kb .. +7
            h8 r2 = *(const h8*)(ab + 16);   // K = 32c+kb+16 .. +23
            v16h a = __builtin_shufflevector(r1, r2, 0, 1, 2, 3, 4, 5, 6, 7,
                                             8, 9, 10, 11, 12, 13, 14, 15);
            d0 = __builtin_amdgcn_wmma_f32_16x16x32_f16(
                false, a, false, bf[0][c].v, (short)0, d0, false, false);
            d1 = __builtin_amdgcn_wmma_f32_16x16x32_f16(
                false, a, false, bf[1][c].v, (short)0, d1, false, false);
        }

        // pure add/add/max fold: -INF penalty kills masked-out (j,row) pairs
#pragma unroll
        for (int r = 0; r < 8; ++r) {
            const float t0 = d0[r] + pen8[r];
            const float t1 = d1[r] + pen8[r];
            acc0[r] = fmaxf(acc0[r], t0 + m2a);
            acc1[r] = fmaxf(acc1[r], t1 + m2b);
        }
        __syncthreads();
    }

    // --- store partial 16x128 max tile -------------------------------------
    float* pb2 = part + (size_t)blk * (IT * MIDD);
#pragma unroll
    for (int r = 0; r < 8; ++r) {
        pb2[(r + hioff) * MIDD + mlane]      = acc0[r];
        pb2[(r + hioff) * MIDD + mlane + 16] = acc1[r];
    }
}

// ---------------------------------------------------------------------------
// Kernel 3: reduce j-chunk partials, clamp, add c1, epilogue GEMM, ReLU.
// One block of 128 threads per (b, i) row.
// ---------------------------------------------------------------------------
__global__ __launch_bounds__(128) void pgn_out_kernel(
    const float* __restrict__ part, const float* __restrict__ c1,
    const float* __restrict__ o1, const float* __restrict__ W_o2,
    const float* __restrict__ b_o2, float* __restrict__ out)
{
    const int b  = blockIdx.x / NN;
    const int i  = blockIdx.x % NN;
    const int m  = threadIdx.x;
    const int it = i >> 4, ir = i & 15;

    __shared__ float agg[MIDD];

    float v = -INFINITY;
#pragma unroll
    for (int q = 0; q < NJ; ++q)
        v = fmaxf(v, part[(((size_t)(b * NIT + it) * NJ + q) * IT + ir) * MIDD + m]);

    const size_t bi = (size_t)b * NN + i;
    agg[m] = fmaxf(-1000000.0f, c1[bi * MIDD + m] + v);
    __syncthreads();

    float s = o1[bi * MIDD + m] + b_o2[m];
#pragma unroll 8
    for (int k = 0; k < MIDD; ++k)
        s = fmaf(agg[k], W_o2[k * MIDD + m], s);

    out[bi * MIDD + m] = fmaxf(s, 0.0f);
}

// ---------------------------------------------------------------------------
extern "C" void kernel_launch(void* const* d_in, const int* in_sizes, int n_in,
                              void* d_out, int out_size, void* d_ws, size_t ws_size,
                              hipStream_t stream) {
    (void)in_sizes; (void)n_in; (void)out_size; (void)ws_size;

    const float* node   = (const float*)d_in[0];
    const float* edge   = (const float*)d_in[1];
    const float* graph  = (const float*)d_in[2];
    const int*   adj    = (const int*)  d_in[3];
    const float* hidden = (const float*)d_in[4];
    const float* W_m1 = (const float*)d_in[5];  const float* b_m1 = (const float*)d_in[6];
    const float* W_m2 = (const float*)d_in[7];  const float* b_m2 = (const float*)d_in[8];
    const float* W_me = (const float*)d_in[9];  const float* b_me = (const float*)d_in[10];
    const float* W_mg = (const float*)d_in[11]; const float* b_mg = (const float*)d_in[12];
    const float* W_o1 = (const float*)d_in[13]; const float* b_o1 = (const float*)d_in[14];
    const float* W_o2 = (const float*)d_in[15]; const float* b_o2 = (const float*)d_in[16];
    float* out = (float*)d_out;

    // workspace layout (floats): m2 | c1 | o1 | partials   (~7 MB total)
    float* ws   = (float*)d_ws;
    float* m2   = ws;
    float* c1   = ws + (size_t)BB * NN * MIDD;
    float* o1   = ws + (size_t)2 * BB * NN * MIDD;
    float* part = ws + (size_t)3 * BB * NN * MIDD;  // B*NIT*NJ*16*128 floats

    pgn_pre_kernel<<<(BB * NN * MIDD) / 256, 256, 0, stream>>>(
        node, hidden, graph, W_m1, b_m1, W_m2, b_m2, W_mg, b_mg, b_me,
        W_o1, b_o1, m2, c1, o1);

    pgn_edge_wmma_kernel<<<BB * NIT * NJ, 128, 0, stream>>>(
        edge, adj, W_me, m2, part);

    pgn_out_kernel<<<BB * NN, 128, 0, stream>>>(
        part, c1, o1, W_o2, b_o2, out);
}
